// GVPEncoder_2113123910149
// MI455X (gfx1250) — compile-verified, matrix operand order checked by reading, not run on hardware
//
#include <hip/hip_runtime.h>
#include <math.h>

// ---------------------------------------------------------------------------
// CDNA5 (gfx1250) GVP-GNN encoder.
// Heavy per-edge / per-node scalar GEMMs run on v_wmma_f32_16x16x32_bf16.
// A-panels are written to LDS directly in the ISA fragment layout
// (CDNA5 ISA 16-bit A 16x32 table), weights are pre-swizzled into
// B-fragment order, so every WMMA operand load is one contiguous 32B load.
// ---------------------------------------------------------------------------

typedef __attribute__((ext_vector_type(16))) __bf16 v16bf;
typedef __attribute__((ext_vector_type(8)))  float  v8f;

#define DEV __device__ __forceinline__

DEV v8f wmma_bf16(v16bf a, v16bf b, v8f c) {
  // 8 args: (neg_a, A, neg_b, B, c_mod, C, reuse_a, reuse_b)
  return __builtin_amdgcn_wmma_f32_16x16x32_bf16(false, a, false, b, (short)0, c,
                                                 false, false);
}

// Store scalar (m,k) element into an A-panel laid out as WMMA bf16 A fragments:
// lane = m + 16*((k&31)>>3 & 1), element j = ((k&31)>>4)*8 | (k&7), chunk kc=k>>5.
DEV void a_store(__bf16* tileBase, int m, int k, float val) {
  int kc = k >> 5, f = k & 31;
  int lane = m + (((f >> 3) & 1) << 4);
  int j = ((f >> 4) << 3) | (f & 7);
  tileBase[(kc * 32 + lane) * 16 + j] = (__bf16)val;
}

DEV float sigmoidf_(float x) { return 1.0f / (1.0f + __expf(-x)); }

DEV void atomAddF(float* p, float v) { unsafeAtomicAdd(p, v); }

// ---------------------------------------------------------------------------
__global__ void k_zero(float* p, int n) {
  int i = blockIdx.x * blockDim.x + threadIdx.x;
  if (i < n) p[i] = 0.0f;
}

// Row-major f32 weight [K x N]  ->  bf16 B-fragment panels:
// panel[((kc*nTiles+nt)*32 + lane)*16 + j], mirror of the A layout with M->N.
__global__ void k_swizzle(const float* __restrict__ W, __bf16* __restrict__ out,
                          int K, int N, int kChunks, int nTiles) {
  int idx = blockIdx.x * blockDim.x + threadIdx.x;
  int total = kChunks * nTiles * 512;
  if (idx >= total) return;
  int j = idx & 15, lane = (idx >> 4) & 31, t = idx >> 9;
  int nt = t % nTiles, kc = t / nTiles;
  int half = lane >> 4, nloc = lane & 15;
  int f = ((j >> 3) << 4) | (half << 3) | (j & 7);
  int k = kc * 32 + f, n = nt * 16 + nloc;
  float w = (k < K && n < N) ? W[(size_t)k * N + n] : 0.0f;
  out[idx] = (__bf16)w;
}

// ---------------------------------------------------------------------------
// Node embedding: LN(6 scalars, 3 vec channels) + GVP (6,3)->(100,16), no acts.
__global__ void k_embed_nodes(const float* __restrict__ ns, const float* __restrict__ nv,
                              const float* lng, const float* lnb,
                              const float* wh, const float* wsw, const float* wsb,
                              const float* wvv,
                              float* s_out, float* v_out, int N) {
  __shared__ float wsL[22 * 100];
  __shared__ float whL[48];
  __shared__ float wvL[256];
  for (int i = threadIdx.x; i < 2200; i += blockDim.x) wsL[i] = wsw[i];
  for (int i = threadIdx.x; i < 48; i += blockDim.x) whL[i] = wh[i];
  for (int i = threadIdx.x; i < 256; i += blockDim.x) wvL[i] = wvv[i];
  __syncthreads();
  int node = blockIdx.x * blockDim.x + threadIdx.x;
  if (node >= N) return;
  float sr[6], sum = 0.f;
  for (int k = 0; k < 6; ++k) { sr[k] = ns[(size_t)node * 6 + k]; sum += sr[k]; }
  float mu = sum / 6.f, var = 0.f;
  for (int k = 0; k < 6; ++k) { float d = sr[k] - mu; var += d * d; }
  float rstd = rsqrtf(var / 6.f + 1e-5f);
  for (int k = 0; k < 6; ++k) sr[k] = (sr[k] - mu) * rstd * lng[k] + lnb[k];
  float vr[9], msq = 0.f;
  for (int c = 0; c < 3; ++c) {
    float n2 = 0.f;
    for (int d = 0; d < 3; ++d) {
      vr[c * 3 + d] = nv[(size_t)node * 9 + c * 3 + d];
      n2 += vr[c * 3 + d] * vr[c * 3 + d];
    }
    msq += fmaxf(n2, 1e-8f);
  }
  float vscale = rsqrtf(msq / 3.f);
  for (int q = 0; q < 9; ++q) vr[q] *= vscale;
  float vha[16][3];
  for (int h = 0; h < 16; ++h) { vha[h][0] = 0.f; vha[h][1] = 0.f; vha[h][2] = 0.f; }
  for (int i = 0; i < 3; ++i)
    for (int h = 0; h < 16; ++h) {
      float w = whL[i * 16 + h];
      vha[h][0] += vr[i * 3 + 0] * w;
      vha[h][1] += vr[i * 3 + 1] * w;
      vha[h][2] += vr[i * 3 + 2] * w;
    }
  float vn[16];
  for (int h = 0; h < 16; ++h)
    vn[h] = sqrtf(fmaxf(vha[h][0] * vha[h][0] + vha[h][1] * vha[h][1] +
                        vha[h][2] * vha[h][2], 1e-8f));
  for (int o = 0; o < 100; ++o) {
    float acc = wsb[o];
    for (int k = 0; k < 6; ++k) acc += sr[k] * wsL[k * 100 + o];
    for (int k = 0; k < 16; ++k) acc += vn[k] * wsL[(6 + k) * 100 + o];
    s_out[(size_t)node * 100 + o] = acc;
  }
  for (int o = 0; o < 16; ++o) {
    float a0 = 0.f, a1 = 0.f, a2 = 0.f;
    for (int h = 0; h < 16; ++h) {
      float w = wvL[h * 16 + o];
      a0 += vha[h][0] * w; a1 += vha[h][1] * w; a2 += vha[h][2] * w;
    }
    v_out[(size_t)node * 48 + o * 3 + 0] = a0;
    v_out[(size_t)node * 48 + o * 3 + 1] = a1;
    v_out[(size_t)node * 48 + o * 3 + 2] = a2;
  }
}

// Edge embedding: LN(32 scalars, 1 vec channel) + GVP (32,1)->(32,1), no acts.
__global__ void k_embed_edges(const float* __restrict__ esi, const float* __restrict__ evi,
                              const float* lng, const float* lnb,
                              const float* wh, const float* wsw, const float* wsb,
                              const float* wvv,
                              float* es_out, float* ev_out, int E) {
  __shared__ float wsL[33 * 32];
  for (int i = threadIdx.x; i < 33 * 32; i += blockDim.x) wsL[i] = wsw[i];
  __syncthreads();
  int e = blockIdx.x * blockDim.x + threadIdx.x;
  if (e >= E) return;
  float sr[32], sum = 0.f;
  for (int k = 0; k < 32; ++k) { sr[k] = esi[(size_t)e * 32 + k]; sum += sr[k]; }
  float mu = sum / 32.f, var = 0.f;
  for (int k = 0; k < 32; ++k) { float d = sr[k] - mu; var += d * d; }
  float rstd = rsqrtf(var / 32.f + 1e-5f);
  for (int k = 0; k < 32; ++k) sr[k] = (sr[k] - mu) * rstd * lng[k] + lnb[k];
  float v0 = evi[(size_t)e * 3], v1 = evi[(size_t)e * 3 + 1], v2 = evi[(size_t)e * 3 + 2];
  float n2 = fmaxf(v0 * v0 + v1 * v1 + v2 * v2, 1e-8f);
  float vs = rsqrtf(n2);  // mean over 1 channel
  v0 *= vs; v1 *= vs; v2 *= vs;
  float w00 = wh[0];
  float h0 = v0 * w00, h1 = v1 * w00, h2 = v2 * w00;
  float vn = sqrtf(fmaxf(h0 * h0 + h1 * h1 + h2 * h2, 1e-8f));
  for (int o = 0; o < 32; ++o) {
    float acc = wsb[o];
    for (int k = 0; k < 32; ++k) acc += sr[k] * wsL[k * 32 + o];
    acc += vn * wsL[32 * 32 + o];
    es_out[(size_t)e * 32 + o] = acc;
  }
  float wv0 = wvv[0];
  ev_out[(size_t)e * 3 + 0] = h0 * wv0;
  ev_out[(size_t)e * 3 + 1] = h1 * wv0;
  ev_out[(size_t)e * 3 + 2] = h2 * wv0;
}

// ---------------------------------------------------------------------------
// Conv: per-edge 3-GVP message stack, 16 edges per single-wave workgroup.
// Scalar paths: WMMA GEMMs [16x288]@[288x112], 2x [16x128]@[128x112].
// Vector paths (33->33, 33->16, 16->16, 16->16 on 3-vectors): VALU from LDS.
// Aggregation: f32 atomics into agg_s/agg_v/cnt (L2-resident).
__global__ void __launch_bounds__(32) k_conv(
    const float* __restrict__ s, const float* __restrict__ v,
    const float* __restrict__ es, const float* __restrict__ ev,
    const int* __restrict__ ei,
    const float* wh0, const float* wv0, const float* wsb0, const v16bf* __restrict__ w0p,
    const float* wh1, const float* wv1, const float* wsb1, const v16bf* __restrict__ w1p,
    const float* wh2, const float* wv2, const float* wsb2, const v16bf* __restrict__ w2p,
    float* agg_s, float* agg_v, float* cnt, int E) {
  __shared__ v16bf aPanel[9 * 32];    // A0: 288-pad (9 chunks); reused for A1/A2 (128-pad)
  __shared__ float vhb[16][33][3];
  __shared__ float mvb[16][33][3];
  __shared__ float wt[1089];
  __shared__ int dstIds[16];
  const int tid = threadIdx.x, lane = tid;
  const int e = tid & 15, p = tid >> 4, m = e;
  __bf16* aw = (__bf16*)aPanel;

  for (int i = tid; i < 1089; i += 32) wt[i] = wh0[i];  // wh0: 33x33

  long ebase = (long)blockIdx.x * 16 + e;
  int eid = (ebase < E) ? (int)ebase : (E - 1);
  int srcN = ei[eid], dstN = ei[(size_t)E + eid];
  if (p == 0) dstIds[e] = (ebase < E) ? dstN : -1;
  {
    const float4* s4 = (const float4*)(s + (size_t)srcN * 100);
    for (int q = p; q < 25; q += 2) {
      float4 x = s4[q];
      a_store(aw, m, 4 * q + 0, x.x); a_store(aw, m, 4 * q + 1, x.y);
      a_store(aw, m, 4 * q + 2, x.z); a_store(aw, m, 4 * q + 3, x.w);
    }
    const float4* e4 = (const float4*)(es + (size_t)eid * 32);
    for (int q = p; q < 8; q += 2) {
      float4 x = e4[q];
      a_store(aw, m, 100 + 4 * q + 0, x.x); a_store(aw, m, 100 + 4 * q + 1, x.y);
      a_store(aw, m, 100 + 4 * q + 2, x.z); a_store(aw, m, 100 + 4 * q + 3, x.w);
    }
    const float4* d4 = (const float4*)(s + (size_t)dstN * 100);
    for (int q = p; q < 25; q += 2) {
      float4 x = d4[q];
      a_store(aw, m, 132 + 4 * q + 0, x.x); a_store(aw, m, 132 + 4 * q + 1, x.y);
      a_store(aw, m, 132 + 4 * q + 2, x.z); a_store(aw, m, 132 + 4 * q + 3, x.w);
    }
    for (int k = 265 + p; k < 288; k += 2) a_store(aw, m, k, 0.f);
    const float* vs = v + (size_t)srcN * 48;
    for (int i = p; i < 16; i += 2) {
      mvb[e][i][0] = vs[i * 3]; mvb[e][i][1] = vs[i * 3 + 1]; mvb[e][i][2] = vs[i * 3 + 2];
    }
    if (p == 0) {
      mvb[e][16][0] = ev[(size_t)eid * 3];
      mvb[e][16][1] = ev[(size_t)eid * 3 + 1];
      mvb[e][16][2] = ev[(size_t)eid * 3 + 2];
    }
    const float* vd = v + (size_t)dstN * 48;
    for (int i = p; i < 16; i += 2) {
      mvb[e][17 + i][0] = vd[i * 3]; mvb[e][17 + i][1] = vd[i * 3 + 1];
      mvb[e][17 + i][2] = vd[i * 3 + 2];
    }
  }
  __syncthreads();
  // vh0 = mv @ wh0 (33->33); vn -> A0[232+h]
  {
    float vha[17][3];
#pragma unroll
    for (int x = 0; x < 17; ++x) { vha[x][0] = 0.f; vha[x][1] = 0.f; vha[x][2] = 0.f; }
    for (int i = 0; i < 33; ++i) {
      float m0 = mvb[e][i][0], m1 = mvb[e][i][1], m2 = mvb[e][i][2];
#pragma unroll
      for (int x = 0; x < 17; ++x) {
        int h = p + 2 * x;
        if (h < 33) {
          float w = wt[i * 33 + h];
          vha[x][0] += m0 * w; vha[x][1] += m1 * w; vha[x][2] += m2 * w;
        }
      }
    }
#pragma unroll
    for (int x = 0; x < 17; ++x) {
      int h = p + 2 * x;
      if (h < 33) {
        vhb[e][h][0] = vha[x][0]; vhb[e][h][1] = vha[x][1]; vhb[e][h][2] = vha[x][2];
        float n2 = vha[x][0] * vha[x][0] + vha[x][1] * vha[x][1] + vha[x][2] * vha[x][2];
        a_store(aw, m, 232 + h, sqrtf(fmaxf(n2, 1e-8f)));
      }
    }
  }
  __syncthreads();
  // GEMM0: s0 = A0[16x288] @ W0[288x112]
  v8f acc0[7] = {};
  for (int kc = 0; kc < 9; ++kc) {
    v16bf a = aPanel[kc * 32 + lane];
#pragma unroll
    for (int nt = 0; nt < 7; ++nt)
      acc0[nt] = wmma_bf16(a, w0p[(kc * 7 + nt) * 32 + lane], acc0[nt]);
  }
  for (int i = tid; i < 528; i += 32) wt[i] = wv0[i];        // wv0: 33x16
  for (int i = tid; i < 256; i += 32) wt[528 + i] = wh1[i];  // wh1: 16x16
  __syncthreads();
  // write s0 (bias+relu) into A1[k<100]; v0 = sigmoid-gated vh0 @ wv0 -> mvb
  {
    int nloc = lane & 15, half = lane >> 4;
#pragma unroll
    for (int nt = 0; nt < 7; ++nt) {
      int col = nt * 16 + nloc;
      if (col < 100) {
        float bias = wsb0[col];
#pragma unroll
        for (int r = 0; r < 8; ++r)
          a_store(aw, half * 8 + r, col, fmaxf(acc0[nt][r] + bias, 0.f));
      }
    }
#pragma unroll
    for (int x = 0; x < 8; ++x) {
      int o = p + 2 * x;
      float a0 = 0.f, a1 = 0.f, a2 = 0.f;
      for (int h = 0; h < 33; ++h) {
        float w = wt[h * 16 + o];
        a0 += vhb[e][h][0] * w; a1 += vhb[e][h][1] * w; a2 += vhb[e][h][2] * w;
      }
      float n = sqrtf(fmaxf(a0 * a0 + a1 * a1 + a2 * a2, 1e-8f));
      float sg = sigmoidf_(n);
      mvb[e][o][0] = a0 * sg; mvb[e][o][1] = a1 * sg; mvb[e][o][2] = a2 * sg;
    }
  }
  __syncthreads();
  // vh1 = v0 @ wh1 (16->16); vn1 -> A1[100+h]; zero pad [116,128)
  {
#pragma unroll
    for (int x = 0; x < 8; ++x) {
      int h = p + 2 * x;
      float a0 = 0.f, a1 = 0.f, a2 = 0.f;
      for (int o = 0; o < 16; ++o) {
        float w = wt[528 + o * 16 + h];
        a0 += mvb[e][o][0] * w; a1 += mvb[e][o][1] * w; a2 += mvb[e][o][2] * w;
      }
      vhb[e][h][0] = a0; vhb[e][h][1] = a1; vhb[e][h][2] = a2;
      a_store(aw, m, 100 + h, sqrtf(fmaxf(a0 * a0 + a1 * a1 + a2 * a2, 1e-8f)));
    }
    for (int k = 116 + p; k < 128; k += 2) a_store(aw, m, k, 0.f);
  }
  __syncthreads();
  // GEMM1: s1 = A1[16x128] @ W1[128x112]
  v8f acc1[7] = {};
  for (int kc = 0; kc < 4; ++kc) {
    v16bf a = aPanel[kc * 32 + lane];
#pragma unroll
    for (int nt = 0; nt < 7; ++nt)
      acc1[nt] = wmma_bf16(a, w1p[(kc * 7 + nt) * 32 + lane], acc1[nt]);
  }
  for (int i = tid; i < 256; i += 32) wt[i] = wv1[i];        // wv1: 16x16
  for (int i = tid; i < 256; i += 32) wt[256 + i] = wh2[i];  // wh2: 16x16
  __syncthreads();
  // write s1 (bias+relu) into A2[k<100]; v1 = sigmoid-gated vh1 @ wv1 -> mvb
  {
    int nloc = lane & 15, half = lane >> 4;
#pragma unroll
    for (int nt = 0; nt < 7; ++nt) {
      int col = nt * 16 + nloc;
      if (col < 100) {
        float bias = wsb1[col];
#pragma unroll
        for (int r = 0; r < 8; ++r)
          a_store(aw, half * 8 + r, col, fmaxf(acc1[nt][r] + bias, 0.f));
      }
    }
#pragma unroll
    for (int x = 0; x < 8; ++x) {
      int o = p + 2 * x;
      float a0 = 0.f, a1 = 0.f, a2 = 0.f;
      for (int h = 0; h < 16; ++h) {
        float w = wt[h * 16 + o];
        a0 += vhb[e][h][0] * w; a1 += vhb[e][h][1] * w; a2 += vhb[e][h][2] * w;
      }
      float n = sqrtf(fmaxf(a0 * a0 + a1 * a1 + a2 * a2, 1e-8f));
      float sg = sigmoidf_(n);
      mvb[e][o][0] = a0 * sg; mvb[e][o][1] = a1 * sg; mvb[e][o][2] = a2 * sg;
    }
  }
  __syncthreads();
  // vh2 = v1 @ wh2; vn2 -> A2[100+h] (pad [116,128) still zero)
  {
#pragma unroll
    for (int x = 0; x < 8; ++x) {
      int h = p + 2 * x;
      float a0 = 0.f, a1 = 0.f, a2 = 0.f;
      for (int o = 0; o < 16; ++o) {
        float w = wt[256 + o * 16 + h];
        a0 += mvb[e][o][0] * w; a1 += mvb[e][o][1] * w; a2 += mvb[e][o][2] * w;
      }
      vhb[e][h][0] = a0; vhb[e][h][1] = a1; vhb[e][h][2] = a2;
      a_store(aw, m, 100 + h, sqrtf(fmaxf(a0 * a0 + a1 * a1 + a2 * a2, 1e-8f)));
    }
  }
  __syncthreads();
  // GEMM2: s2 = A2[16x128] @ W2[128x112] (no relu)
  v8f acc2[7] = {};
  for (int kc = 0; kc < 4; ++kc) {
    v16bf a = aPanel[kc * 32 + lane];
#pragma unroll
    for (int nt = 0; nt < 7; ++nt)
      acc2[nt] = wmma_bf16(a, w2p[(kc * 7 + nt) * 32 + lane], acc2[nt]);
  }
  for (int i = tid; i < 256; i += 32) wt[i] = wv2[i];  // wv2: 16x16
  __syncthreads();
  // scatter (mean aggregation numerators + counts)
  {
    int nloc = lane & 15, half = lane >> 4;
#pragma unroll
    for (int nt = 0; nt < 7; ++nt) {
      int col = nt * 16 + nloc;
      if (col < 100) {
        float bias = wsb2[col];
#pragma unroll
        for (int r = 0; r < 8; ++r) {
          int d = dstIds[half * 8 + r];
          if (d >= 0) atomAddF(&agg_s[(size_t)d * 100 + col], acc2[nt][r] + bias);
        }
      }
    }
    int d = dstIds[e];
    if (d >= 0) {
#pragma unroll
      for (int x = 0; x < 8; ++x) {
        int o = p + 2 * x;
        float a0 = 0.f, a1 = 0.f, a2 = 0.f;
        for (int h = 0; h < 16; ++h) {
          float w = wt[h * 16 + o];
          a0 += vhb[e][h][0] * w; a1 += vhb[e][h][1] * w; a2 += vhb[e][h][2] * w;
        }
        atomAddF(&agg_v[(size_t)d * 48 + o * 3 + 0], a0);
        atomAddF(&agg_v[(size_t)d * 48 + o * 3 + 1], a1);
        atomAddF(&agg_v[(size_t)d * 48 + o * 3 + 2], a2);
      }
      if (p == 0) atomAddF(&cnt[d], 1.0f);
    }
  }
}

// ---------------------------------------------------------------------------
// residual + LayerNorm (100 scalars, 16 vec channels). cnt!=nullptr => mean agg.
__global__ void k_residual_ln(float* s, float* v, const float* __restrict__ ds,
                              const float* __restrict__ dv, const float* cnt,
                              const float* g, const float* b, int N) {
  int i = blockIdx.x * blockDim.x + threadIdx.x;
  if (i >= N) return;
  float ic = 1.f;
  if (cnt) ic = 1.f / fmaxf(cnt[i], 1.f);
  float sum = 0.f, sq = 0.f;
  for (int k = 0; k < 100; ++k) {
    float x = s[(size_t)i * 100 + k] + ds[(size_t)i * 100 + k] * ic;
    sum += x; sq += x * x;
  }
  float mu = sum * 0.01f;
  float var = sq * 0.01f - mu * mu;
  float rstd = rsqrtf(fmaxf(var, 0.f) + 1e-5f);
  for (int k = 0; k < 100; ++k) {
    float x = s[(size_t)i * 100 + k] + ds[(size_t)i * 100 + k] * ic;
    s[(size_t)i * 100 + k] = (x - mu) * rstd * g[k] + b[k];
  }
  float msq = 0.f;
  for (int c = 0; c < 16; ++c) {
    float n2 = 0.f;
    for (int d = 0; d < 3; ++d) {
      float x = v[(size_t)i * 48 + c * 3 + d] + dv[(size_t)i * 48 + c * 3 + d] * ic;
      n2 += x * x;
    }
    msq += fmaxf(n2, 1e-8f);
  }
  float vs = rsqrtf(msq / 16.f);
  for (int q = 0; q < 48; ++q) {
    float x = v[(size_t)i * 48 + q] + dv[(size_t)i * 48 + q] * ic;
    v[(size_t)i * 48 + q] = x * vs;
  }
}

// ---------------------------------------------------------------------------
// Feed-forward: GVP (100,16)->(400,32) acts, then (400,32)->(100,16) no acts.
// 16 nodes per single-wave WG; WMMA [16x160]@[160x400], [16x448]@[448x112].
__global__ void __launch_bounds__(32) k_ff(
    const float* __restrict__ s, const float* __restrict__ v,
    const float* wh0, const float* wv0, const float* wsb0, const v16bf* __restrict__ w0p,
    const float* wh1, const float* wv1, const float* wsb1, const v16bf* __restrict__ w1p,
    float* fs, float* fv, int N) {
  __shared__ v16bf aP1[5 * 32];    // A1: [s(100), vn(32)] pad 160
  __shared__ v16bf aP2[14 * 32];   // A2: [fs(400), vn2(32)] pad 448
  __shared__ float vhb[16][32][3];
  __shared__ float mvb[16][32][3];
  __shared__ float wt[1024];
  const int tid = threadIdx.x, lane = tid;
  const int e = tid & 15, p = tid >> 4, m = e;
  __bf16* aw1 = (__bf16*)aP1;
  __bf16* aw2 = (__bf16*)aP2;
  long nbase = (long)blockIdx.x * 16 + e;
  int node = (nbase < N) ? (int)nbase : (N - 1);
  bool valid = (nbase < N);

  for (int i = tid; i < 512; i += 32) wt[i] = wh0[i];  // wh0: 16x32
  {
    const float4* s4 = (const float4*)(s + (size_t)node * 100);
    for (int q = p; q < 25; q += 2) {
      float4 x = s4[q];
      a_store(aw1, m, 4 * q + 0, x.x); a_store(aw1, m, 4 * q + 1, x.y);
      a_store(aw1, m, 4 * q + 2, x.z); a_store(aw1, m, 4 * q + 3, x.w);
    }
    for (int k = 132 + p; k < 160; k += 2) a_store(aw1, m, k, 0.f);
    const float* vp = v + (size_t)node * 48;
    for (int i = p; i < 16; i += 2) {
      mvb[e][i][0] = vp[i * 3]; mvb[e][i][1] = vp[i * 3 + 1]; mvb[e][i][2] = vp[i * 3 + 2];
    }
  }
  __syncthreads();
  // vh = v @ wh0 (16->32); vn -> A1[100+h]
#pragma unroll
  for (int x = 0; x < 16; ++x) {
    int h = p + 2 * x;
    float a0 = 0.f, a1 = 0.f, a2 = 0.f;
    for (int i = 0; i < 16; ++i) {
      float w = wt[i * 32 + h];
      a0 += mvb[e][i][0] * w; a1 += mvb[e][i][1] * w; a2 += mvb[e][i][2] * w;
    }
    vhb[e][h][0] = a0; vhb[e][h][1] = a1; vhb[e][h][2] = a2;
    a_store(aw1, m, 100 + h, sqrtf(fmaxf(a0 * a0 + a1 * a1 + a2 * a2, 1e-8f)));
  }
  __syncthreads();
  // GEMM-A: fs = relu(A1[16x160] @ W0[160x400] + b) -> A2[k<400]
  {
    v16bf aF[5];
#pragma unroll
    for (int kc = 0; kc < 5; ++kc) aF[kc] = aP1[kc * 32 + lane];
    int nloc = lane & 15, half = lane >> 4;
    for (int nt = 0; nt < 25; ++nt) {
      v8f acc = {};
#pragma unroll
      for (int kc = 0; kc < 5; ++kc)
        acc = wmma_bf16(aF[kc], w0p[(kc * 25 + nt) * 32 + lane], acc);
      int col = nt * 16 + nloc;  // always < 400
      float bias = wsb0[col];
#pragma unroll
      for (int r = 0; r < 8; ++r)
        a_store(aw2, half * 8 + r, col, fmaxf(acc[r] + bias, 0.f));
    }
  }
  for (int i = tid; i < 1024; i += 32) wt[i] = wv0[i];  // wv0: 32x32
  __syncthreads();
  // v1 = sigmoid-gated vh @ wv0 (32->32) -> mvb
#pragma unroll
  for (int x = 0; x < 16; ++x) {
    int o = p + 2 * x;
    float a0 = 0.f, a1 = 0.f, a2 = 0.f;
    for (int h = 0; h < 32; ++h) {
      float w = wt[h * 32 + o];
      a0 += vhb[e][h][0] * w; a1 += vhb[e][h][1] * w; a2 += vhb[e][h][2] * w;
    }
    float n = sqrtf(fmaxf(a0 * a0 + a1 * a1 + a2 * a2, 1e-8f));
    float sg = sigmoidf_(n);
    mvb[e][o][0] = a0 * sg; mvb[e][o][1] = a1 * sg; mvb[e][o][2] = a2 * sg;
  }
  __syncthreads();
  for (int i = tid; i < 1024; i += 32) wt[i] = wh1[i];  // wh1: 32x32
  __syncthreads();
  // vh2 = v1 @ wh1; vn2 -> A2[400+h]; pad [432,448)
#pragma unroll
  for (int x = 0; x < 16; ++x) {
    int h = p + 2 * x;
    float a0 = 0.f, a1 = 0.f, a2 = 0.f;
    for (int o = 0; o < 32; ++o) {
      float w = wt[o * 32 + h];
      a0 += mvb[e][o][0] * w; a1 += mvb[e][o][1] * w; a2 += mvb[e][o][2] * w;
    }
    vhb[e][h][0] = a0; vhb[e][h][1] = a1; vhb[e][h][2] = a2;
    a_store(aw2, m, 400 + h, sqrtf(fmaxf(a0 * a0 + a1 * a1 + a2 * a2, 1e-8f)));
  }
  for (int k = 432 + p; k < 448; k += 2) a_store(aw2, m, k, 0.f);
  __syncthreads();
  // GEMM-B: out = A2[16x448] @ W1[448x112] + b (no relu)
  v8f accB[7] = {};
  for (int kc = 0; kc < 14; ++kc) {
    v16bf a = aP2[kc * 32 + lane];
#pragma unroll
    for (int nt = 0; nt < 7; ++nt)
      accB[nt] = wmma_bf16(a, w1p[(kc * 7 + nt) * 32 + lane], accB[nt]);
  }
  for (int i = tid; i < 512; i += 32) wt[i] = wv1[i];  // wv1: 32x16
  __syncthreads();
  {
    int nloc = lane & 15, half = lane >> 4;
#pragma unroll
    for (int nt = 0; nt < 7; ++nt) {
      int col = nt * 16 + nloc;
      if (col < 100) {
        float bias = wsb1[col];
#pragma unroll
        for (int r = 0; r < 8; ++r) {
          long nb = (long)blockIdx.x * 16 + half * 8 + r;
          if (nb < N) fs[(size_t)nb * 100 + col] = accB[nt][r] + bias;
        }
      }
    }
    if (valid) {
#pragma unroll
      for (int x = 0; x < 8; ++x) {
        int o = p + 2 * x;
        float a0 = 0.f, a1 = 0.f, a2 = 0.f;
        for (int h = 0; h < 32; ++h) {
          float w = wt[h * 16 + o];
          a0 += vhb[e][h][0] * w; a1 += vhb[e][h][1] * w; a2 += vhb[e][h][2] * w;
        }
        fv[(size_t)node * 48 + o * 3 + 0] = a0;
        fv[(size_t)node * 48 + o * 3 + 1] = a1;
        fv[(size_t)node * 48 + o * 3 + 2] = a2;
      }
    }
  }
}

// ---------------------------------------------------------------------------
// Output: LN + GVP (100,16)->(100,0) with relu; write res and pool numerators.
__global__ void __launch_bounds__(128) k_out(
    const float* __restrict__ s, const float* __restrict__ v, const int* __restrict__ batch,
    const float* lng, const float* lnb,
    const float* wh, const float* wsw, const float* wsb,
    float* res, float* gsum, float* gcnt, int N) {
  __shared__ float wsL[116 * 100];
  __shared__ float whL[256];
  for (int i = threadIdx.x; i < 11600; i += blockDim.x) wsL[i] = wsw[i];
  for (int i = threadIdx.x; i < 256; i += blockDim.x) whL[i] = wh[i];
  __syncthreads();
  int node = blockIdx.x * blockDim.x + threadIdx.x;
  if (node >= N) return;
  const float* sp = s + (size_t)node * 100;
  const float* vp = v + (size_t)node * 48;
  float sum = 0.f, sq = 0.f;
  for (int k = 0; k < 100; ++k) { float x = sp[k]; sum += x; sq += x * x; }
  float mu = sum * 0.01f, var = sq * 0.01f - mu * mu;
  float rstd = rsqrtf(fmaxf(var, 0.f) + 1e-5f);
  float msq = 0.f;
  for (int c = 0; c < 16; ++c) {
    float n2 = 0.f;
    for (int d = 0; d < 3; ++d) { float x = vp[c * 3 + d]; n2 += x * x; }
    msq += fmaxf(n2, 1e-8f);
  }
  float vscale = rsqrtf(msq / 16.f);
  float vha[16][3];
  for (int h = 0; h < 16; ++h) { vha[h][0] = 0.f; vha[h][1] = 0.f; vha[h][2] = 0.f; }
  for (int i = 0; i < 16; ++i) {
    float x0 = vp[i * 3] * vscale, x1 = vp[i * 3 + 1] * vscale, x2 = vp[i * 3 + 2] * vscale;
    for (int h = 0; h < 16; ++h) {
      float w = whL[i * 16 + h];
      vha[h][0] += x0 * w; vha[h][1] += x1 * w; vha[h][2] += x2 * w;
    }
  }
  float vn[16];
  for (int h = 0; h < 16; ++h)
    vn[h] = sqrtf(fmaxf(vha[h][0] * vha[h][0] + vha[h][1] * vha[h][1] +
                        vha[h][2] * vha[h][2], 1e-8f));
  int bt = batch[node];
  for (int cchunk = 0; cchunk < 4; ++cchunk) {
    float acc[25];
    for (int o = 0; o < 25; ++o) acc[o] = wsb[cchunk * 25 + o];
    for (int k = 0; k < 100; ++k) {
      float sval = (sp[k] - mu) * rstd * lng[k] + lnb[k];
      for (int o = 0; o < 25; ++o) acc[o] += sval * wsL[k * 100 + cchunk * 25 + o];
    }
    for (int k = 0; k < 16; ++k) {
      float x = vn[k];
      for (int o = 0; o < 25; ++o) acc[o] += x * wsL[(100 + k) * 100 + cchunk * 25 + o];
    }
    for (int o = 0; o < 25; ++o) {
      float val = fmaxf(acc[o], 0.f);
      res[(size_t)node * 100 + cchunk * 25 + o] = val;
      atomAddF(&gsum[(size_t)bt * 100 + cchunk * 25 + o], val);
    }
  }
  atomAddF(&gcnt[bt], 1.f);
}

__global__ void k_pool(const float* __restrict__ gsum, const float* __restrict__ gcnt,
                       float* __restrict__ out) {
  int i = blockIdx.x * blockDim.x + threadIdx.x;
  if (i < 64 * 100) out[i] = gsum[i] / fmaxf(gcnt[i / 100], 1.f);
}

// ---------------------------------------------------------------------------
// Param leaf order assumption: JAX pytree (dict keys sorted alphabetically):
// We_gvp{wh,ws_b,ws_w,wv}, We_ln{b,g}, Wout_gvp{wh,ws_b,ws_w}, Wout_ln{b,g},
// Wv_gvp{wh,ws_b,ws_w,wv}, Wv_ln{b,g},
// layers[0..2]{conv[0..2]{wh,ws_b,ws_w,wv}, ff[0..1]{...}, ln0{b,g}, ln1{b,g}}
extern "C" void kernel_launch(void* const* d_in, const int* in_sizes, int n_in,
                              void* d_out, int out_size, void* d_ws, size_t ws_size,
                              hipStream_t stream) {
  (void)n_in; (void)out_size; (void)ws_size;
  const int N = in_sizes[0] / 6;
  const int E = in_sizes[2] / 32;
  const int G = 64;
  auto F = [&](int i) { return (const float*)d_in[i]; };
  const int* ei = (const int*)d_in[4];
  const int* batch = (const int*)d_in[5];

  size_t off = 0;
  auto alloc = [&](size_t bytes) -> void* {
    off = (off + 255) & ~(size_t)255;
    void* p = (char*)d_ws + off;
    off += bytes;
    return p;
  };
  float* s_cur = (float*)alloc((size_t)N * 100 * 4);
  float* v_cur = (float*)alloc((size_t)N * 48 * 4);
  float* esb   = (float*)alloc((size_t)E * 32 * 4);
  float* evb   = (float*)alloc((size_t)E * 3 * 4);
  float* agg_s = (float*)alloc((size_t)N * 100 * 4);
  float* agg_v = (float*)alloc((size_t)N * 48 * 4);
  float* cnt   = (float*)alloc((size_t)N * 4);
  float* fs    = (float*)alloc((size_t)N * 100 * 4);
  float* fv    = (float*)alloc((size_t)N * 48 * 4);
  float* gsum  = (float*)alloc((size_t)G * 100 * 4);
  float* gcnt  = (float*)alloc((size_t)G * 4);
  v16bf* pc0[3]; v16bf* pc1[3]; v16bf* pc2[3]; v16bf* pf0[3]; v16bf* pf1[3];
  for (int L = 0; L < 3; ++L) {
    pc0[L] = (v16bf*)alloc((size_t)9 * 7 * 32 * 32);
    pc1[L] = (v16bf*)alloc((size_t)4 * 7 * 32 * 32);
    pc2[L] = (v16bf*)alloc((size_t)4 * 7 * 32 * 32);
    pf0[L] = (v16bf*)alloc((size_t)5 * 25 * 32 * 32);
    pf1[L] = (v16bf*)alloc((size_t)14 * 7 * 32 * 32);
  }

  float* graph_out = (float*)d_out;
  float* res_out = (float*)d_out + (size_t)G * 100;

  // Pre-swizzle WMMA B panels (bf16 fragment layout) for all layers.
  auto swz = [&](const float* W, v16bf* out, int K, int Ncols, int kc, int nt) {
    int total = kc * nt * 512;
    k_swizzle<<<(total + 255) / 256, 256, 0, stream>>>(W, (__bf16*)out, K, Ncols, kc, nt);
  };
  for (int L = 0; L < 3; ++L) {
    int base = 23 + L * 24;
    swz(F(base + 2),  pc0[L], 265, 100, 9, 7);
    swz(F(base + 6),  pc1[L], 116, 100, 4, 7);
    swz(F(base + 10), pc2[L], 116, 100, 4, 7);
    swz(F(base + 14), pf0[L], 132, 400, 5, 25);
    swz(F(base + 18), pf1[L], 432, 100, 14, 7);
  }

  // Input embeddings.
  k_embed_nodes<<<(N + 255) / 256, 256, 0, stream>>>(
      F(0), F(1), F(22), F(21), F(17), F(19), F(18), F(20), s_cur, v_cur, N);
  k_embed_edges<<<(E + 255) / 256, 256, 0, stream>>>(
      F(2), F(3), F(11), F(10), F(6), F(8), F(7), F(9), esb, evb, E);

  // Message-passing layers.
  for (int L = 0; L < 3; ++L) {
    int base = 23 + L * 24;
    k_zero<<<(N * 100 + 255) / 256, 256, 0, stream>>>(agg_s, N * 100);
    k_zero<<<(N * 48 + 255) / 256, 256, 0, stream>>>(agg_v, N * 48);
    k_zero<<<(N + 255) / 256, 256, 0, stream>>>(cnt, N);
    k_conv<<<(E + 15) / 16, 32, 0, stream>>>(
        s_cur, v_cur, esb, evb, ei,
        F(base + 0), F(base + 3), F(base + 1), pc0[L],
        F(base + 4), F(base + 7), F(base + 5), pc1[L],
        F(base + 8), F(base + 11), F(base + 9), pc2[L],
        agg_s, agg_v, cnt, E);
    k_residual_ln<<<(N + 255) / 256, 256, 0, stream>>>(
        s_cur, v_cur, agg_s, agg_v, cnt, F(base + 21), F(base + 20), N);
    k_ff<<<(N + 15) / 16, 32, 0, stream>>>(
        s_cur, v_cur,
        F(base + 12), F(base + 15), F(base + 13), pf0[L],
        F(base + 16), F(base + 19), F(base + 17), pf1[L],
        fs, fv, N);
    k_residual_ln<<<(N + 255) / 256, 256, 0, stream>>>(
        s_cur, v_cur, fs, fv, nullptr, F(base + 23), F(base + 22), N);
  }

  // Output head + graph mean-pool.
  k_zero<<<(G * 100 + 255) / 256, 256, 0, stream>>>(gsum, G * 100);
  k_zero<<<1, 64, 0, stream>>>(gcnt, G);
  k_out<<<(N + 127) / 128, 128, 0, stream>>>(
      s_cur, v_cur, batch, F(16), F(15), F(12), F(14), F(13), res_out, gsum, gcnt, N);
  k_pool<<<(G * 100 + 255) / 256, 256, 0, stream>>>(gsum, gcnt, graph_out);
}
